// SimpleLgeConvNet_60009283060069
// MI455X (gfx1250) — matrix operand order repaired and checked by reading, not compiled, not run on hardware
//
#include <hip/hip_runtime.h>

typedef _Float16 h16;
typedef __attribute__((ext_vector_type(16))) _Float16 v16h;
typedef __attribute__((ext_vector_type(8)))  _Float16 v8h;
typedef __attribute__((ext_vector_type(8)))  float    v8f;

// Problem constants (from reference)
// DIMS=(8,8,16,16) -> X=16384, B=2, IN_CH=10, D=4, REP=3
// W_IN=6, N_TERMS=9, W_IN_SZ=13, T_W_SZ=109, W_OUT=6
// G1: (654 x 13) @ (13 x 18)  ; G2: (18 x 654) @ (654 x 6)
#define XSITES   16384
#define MG1      654      // 6*109
#define MT_G1    41       // ceil(654/16)
#define KCH_G2   21       // ceil(654/32)
#define PK       672      // padded K for G2 (21*32)
#define PSTRIDE  672      // sP row stride (halves)
#define BSTRIDE  672      // sBT row stride (halves)
// K-index bijection for G2: kk = (j*2 + p)*109 + w   (w fastest -> cheap scatter)

__device__ __forceinline__ size_t xoff(int b, int s, int ch) {
    return ((size_t)(b * XSITES + s) * 10 + (size_t)ch) * 18;
}

struct C33 { float re[9]; float im[9]; };

__device__ __forceinline__ C33 load33(const float* __restrict__ p) {
    C33 m;
    #pragma unroll
    for (int e = 0; e < 9; ++e) { m.re[e] = p[2 * e]; m.im[e] = p[2 * e + 1]; }
    return m;
}

__device__ __forceinline__ C33 dag33(const C33& a) {
    C33 r;
    #pragma unroll
    for (int i = 0; i < 3; ++i)
        #pragma unroll
        for (int j = 0; j < 3; ++j) {
            r.re[i * 3 + j] = a.re[j * 3 + i];
            r.im[i * 3 + j] = -a.im[j * 3 + i];
        }
    return r;
}

__device__ __forceinline__ C33 cmul33(const C33& a, const C33& b) {
    C33 r;
    #pragma unroll
    for (int i = 0; i < 3; ++i)
        #pragma unroll
        for (int k = 0; k < 3; ++k) {
            float re = 0.f, im = 0.f;
            #pragma unroll
            for (int j = 0; j < 3; ++j) {
                re += a.re[i * 3 + j] * b.re[j * 3 + k] - a.im[i * 3 + j] * b.im[j * 3 + k];
                im += a.re[i * 3 + j] * b.im[j * 3 + k] + a.im[i * 3 + j] * b.re[j * 3 + k];
            }
            r.re[i * 3 + k] = re; r.im[i * 3 + k] = im;
        }
    return r;
}

// site = ((d0*8+d1)*16+d2)*16+d3 with dims (8,8,16,16)
__device__ __forceinline__ int neighbor(int site, int axis, int dir) {
    int c[4] = { (site >> 11) & 7, (site >> 8) & 7, (site >> 4) & 15, site & 15 };
    const int dim[4] = { 8, 8, 16, 16 };
    c[axis] = (c[axis] + dir + dim[axis]) % dim[axis];
    return ((c[0] * 8 + c[1]) * 16 + c[2]) * 16 + c[3];
}

// Fill B-side (real-ified complex) panel for channel wB with matrix T.
// sBT layout: [col = k*2+pout][kk = (j*2+pin)*109 + wB], row stride BSTRIDE.
// pin=0 -> (re: Tr, im: Ti) ; pin=1 -> (re: -Ti, im: Tr)
__device__ __forceinline__ void fillB(h16* sBT, int wB, const C33& T) {
    #pragma unroll
    for (int j = 0; j < 3; ++j)
        #pragma unroll
        for (int k = 0; k < 3; ++k) {
            float tr = T.re[j * 3 + k], ti = T.im[j * 3 + k];
            int k0 = (j * 2 + 0) * 109 + wB;
            int k1 = (j * 2 + 1) * 109 + wB;
            sBT[(k * 2 + 0) * BSTRIDE + k0] = (h16)tr;
            sBT[(k * 2 + 1) * BSTRIDE + k0] = (h16)ti;
            sBT[(k * 2 + 0) * BSTRIDE + k1] = (h16)(-ti);
            sBT[(k * 2 + 1) * BSTRIDE + k1] = (h16)tr;
        }
}

// sAhT layout: [row c2=(i*3+j)*2+p][col v], row stride 32
__device__ __forceinline__ void fillA(h16* sAhT, int v, const C33& M) {
    #pragma unroll
    for (int e = 0; e < 9; ++e) {
        sAhT[(2 * e + 0) * 32 + v] = (h16)M.re[e];
        sAhT[(2 * e + 1) * 32 + v] = (h16)M.im[e];
    }
}

// Scatter one G1 C-tile column (this lane's 8 rows) into sP.
// Rows g = mt*16 + base + r are consecutive => running-address update.
template <bool GUARD>
__device__ __forceinline__ void scatterP(h16* sP, const v8f& c, int lane, int Nc, int mt) {
    int p = Nc & 1, ij = Nc >> 1;
    int i = ij / 3, j = ij - 3 * i;
    int g0 = mt * 16 + ((lane < 16) ? 0 : 8);
    int u = g0 / 109, w = g0 - 109 * u;
    int addr = (u * 3 + i) * PSTRIDE + (j * 2 + p) * 109 + w;
    #pragma unroll
    for (int r = 0; r < 8; ++r) {
        if (!GUARD || (g0 + r) < MG1) sP[addr] = (h16)c[r];
        bool wrap = (w == 108);
        addr += wrap ? (3 * PSTRIDE - 108) : 1;
        w = wrap ? 0 : (w + 1);
    }
}

// --- Kernel 0: pack W2 = weight reshaped (654 x 13) into WMMA f16 A-operand
// tile layout (ISA 7.12.2, 16-bit A 16x32). ws[(tile*32+lane)*16 + e].
__global__ void lge_pack_w2(const float* __restrict__ weight, h16* __restrict__ wsA) {
    int tile = blockIdx.x;          // 0..40
    int lane = threadIdx.x;         // 0..31
    int g = tile * 16 + (lane & 15);
    h16 vals[16];
    #pragma unroll
    for (int e = 0; e < 16; ++e) {
        int r = e >> 1, hb = e & 1;
        int K;
        if (lane < 16) K = (r < 4) ? (2 * r + hb)     : (16 + 2 * (r - 4) + hb);
        else           K = (r < 4) ? (8 + 2 * r + hb) : (24 + 2 * (r - 4) + hb);
        float v = 0.f;
        if (g < MG1 && K < 13) {
            int u = g / 109, w = g - 109 * u;
            v = weight[(u * 13 + K) * 109 + w];
        }
        vals[e] = (h16)v;
    }
    h16* dst = wsA + ((size_t)(tile * 32 + lane)) * 16;
    #pragma unroll
    for (int e = 0; e < 16; ++e) dst[e] = vals[e];
}

// --- Kernel 1: one wave (32 lanes) per (b, site).
__global__ void __launch_bounds__(32)
lge_main(const float* __restrict__ x, const h16* __restrict__ wsA, float* __restrict__ out) {
    __shared__ __align__(64) h16 sAhT[32 * 32];        // A-site transposed, zero padded
    __shared__ __align__(64) h16 sBT[16 * BSTRIDE];    // B panel transposed (cols x K)
    __shared__ __align__(64) h16 sP[18 * PSTRIDE];     // P (18 x 654, K-padded to 672)

    const int lane = threadIdx.x;
    const int gidx = blockIdx.x;                // 0..32767
    const int b = gidx >> 14;
    const int site = gidx & (XSITES - 1);

    // ---- zero LDS pads (AhT & BT fully; sP only pad columns) ----
    {
        unsigned int* z1 = (unsigned int*)sAhT;                 // 512 dwords
        for (int i = lane; i < 512; i += 32) z1[i] = 0u;
        unsigned int* z2 = (unsigned int*)sBT;                  // 5376 dwords
        for (int i = lane; i < 5376; i += 32) z2[i] = 0u;
        for (int i = lane; i < 18 * (PK - MG1); i += 32) {      // 18 rows x 18 pad cols
            int r = i / (PK - MG1), c = i - r * (PK - MG1);
            sP[r * PSTRIDE + MG1 + c] = (h16)0.f;
        }
    }
    __syncthreads();

    // ---- transports + operand panel build (55 tasks over 32 lanes) ----
    for (int task = lane; task < 55; task += 32) {
        if (task < 48) {
            int t = task / 6 + 1, ch = task - 6 * (task / 6);   // term 1..8
            int axis = (t - 1) >> 1;
            int o = ((t - 1) & 1) ? 1 : -1;                     // (-1,+1) order
            C33 T;
            if (o > 0) {        // U * W(x+e) * U^dag
                C33 U  = load33(x + xoff(b, site, axis));
                C33 Wn = load33(x + xoff(b, neighbor(site, axis, +1), 4 + ch));
                T = cmul33(cmul33(U, Wn), dag33(U));
            } else {            // U(x-e)^dag * W(x-e) * U(x-e)
                int sm = neighbor(site, axis, -1);
                C33 Us = load33(x + xoff(b, sm, axis));
                C33 Wn = load33(x + xoff(b, sm, 4 + ch));
                T = cmul33(cmul33(dag33(Us), Wn), Us);
            }
            int wB = t * 6 + ch;
            fillB(sBT, wB, T);
            fillB(sBT, wB + 54, dag33(T));
        } else if (task < 54) { // term 0: w itself (+ A-panel rows)
            int ch = task - 48;
            C33 Wm = load33(x + xoff(b, site, 4 + ch));
            fillB(sBT, ch, Wm);
            fillB(sBT, ch + 54, dag33(Wm));
            fillA(sAhT, ch, Wm);
            fillA(sAhT, ch + 6, dag33(Wm));
        } else {                // unit channel
            C33 I = {};
            I.re[0] = I.re[4] = I.re[8] = 1.f;
            fillB(sBT, 108, I);
            fillA(sAhT, 12, I);
        }
    }
    // gauge-link passthrough: out channels 0..3 (72 floats)
    {
        const float* src = x + xoff(b, site, 0);
        float* dst = out + xoff(b, site, 0);
        for (int i = lane; i < 72; i += 32) dst[i] = src[i];
    }
    __syncthreads();

    // ---- G1: P = W2 (654x13) @ A_site (13x18) via v_wmma_f32_16x16x32_f16 ----
    // B-operand (A_site): K-runs contiguous in sAhT rows. Built once, reused 41x.
    v16h bop0, bop1;
    {
        int kb = (lane < 16) ? 0 : 8;
        int c2 = lane & 15;
        v8h lo = *(const v8h*)&sAhT[c2 * 32 + kb];
        v8h hi = *(const v8h*)&sAhT[c2 * 32 + kb + 16];
        bop0 = __builtin_shufflevector(lo, hi, 0,1,2,3,4,5,6,7,8,9,10,11,12,13,14,15);
        c2 = 16 + (lane & 15);
        lo = *(const v8h*)&sAhT[c2 * 32 + kb];
        hi = *(const v8h*)&sAhT[c2 * 32 + kb + 16];
        bop1 = __builtin_shufflevector(lo, hi, 0,1,2,3,4,5,6,7,8,9,10,11,12,13,14,15);
    }
    for (int mt = 0; mt < MT_G1 - 1; ++mt) {        // tiles 0..39: no row guard needed
        v16h aop = *(const v16h*)&wsA[((size_t)(mt * 32 + lane)) * 16];
        __builtin_prefetch(&wsA[((size_t)((mt + 1) * 32 + lane)) * 16], 0, 0);
        v8f c0 = {};
        c0 = __builtin_amdgcn_wmma_f32_16x16x32_f16(false, aop, false, bop0,
                                                    (short)0, c0, false, false);
        scatterP<false>(sP, c0, lane, lane & 15, mt);           // cols 0..15 all valid
        v8f c1 = {};
        c1 = __builtin_amdgcn_wmma_f32_16x16x32_f16(false, aop, false, bop1,
                                                    (short)0, c1, false, false);
        if ((lane & 15) < 2)                                    // cols 16,17 only
            scatterP<false>(sP, c1, lane, 16 + (lane & 15), mt);
    }
    {   // tile 40: rows 654,655 out of range
        const int mt = MT_G1 - 1;
        v16h aop = *(const v16h*)&wsA[((size_t)(mt * 32 + lane)) * 16];
        v8f c0 = {};
        c0 = __builtin_amdgcn_wmma_f32_16x16x32_f16(false, aop, false, bop0,
                                                    (short)0, c0, false, false);
        scatterP<true>(sP, c0, lane, lane & 15, mt);
        v8f c1 = {};
        c1 = __builtin_amdgcn_wmma_f32_16x16x32_f16(false, aop, false, bop1,
                                                    (short)0, c1, false, false);
        if ((lane & 15) < 2)
            scatterP<true>(sP, c1, lane, 16 + (lane & 15), mt);
    }
    __syncthreads();

    // ---- G2: out = P (18x654) @ Bmat (654x6), accumulated over 21 K-chunks ----
    v16h zv;
    #pragma unroll
    for (int e = 0; e < 16; ++e) zv[e] = (h16)0.f;
    #pragma unroll
    for (int mt = 0; mt < 2; ++mt) {
        v8f acc = {};
        int Mrow = mt * 16 + (lane & 15);
        bool mok = (Mrow < 18);
        int a1 = (lane < 16) ? 0 : 8;
        for (int kc = 0; kc < KCH_G2; ++kc) {
            int kb = kc * 32;
            v16h aop = zv;
            if (mok) {
                v8h lo = *(const v8h*)&sP[Mrow * PSTRIDE + kb + a1];
                v8h hi = *(const v8h*)&sP[Mrow * PSTRIDE + kb + 16 + a1];
                aop = __builtin_shufflevector(lo, hi, 0,1,2,3,4,5,6,7,8,9,10,11,12,13,14,15);
            }
            int Nc = lane & 15;
            int bb = kb + ((lane < 16) ? 0 : 8);
            v8h blo = *(const v8h*)&sBT[Nc * BSTRIDE + bb];
            v8h bhi = *(const v8h*)&sBT[Nc * BSTRIDE + bb + 16];
            v16h bop2 = __builtin_shufflevector(blo, bhi, 0,1,2,3,4,5,6,7,8,9,10,11,12,13,14,15);
            acc = __builtin_amdgcn_wmma_f32_16x16x32_f16(false, aop, false, bop2,
                                                         (short)0, acc, false, false);
        }
        int Nc = lane & 15;
        if (Nc < 6) {
            int k = Nc >> 1, p = Nc & 1;
            #pragma unroll
            for (int r = 0; r < 8; ++r) {
                int M = mt * 16 + ((lane < 16) ? r : r + 8);
                if (M < 18) {
                    int u = M / 3, i = M - 3 * u;
                    out[xoff(b, site, 4 + u) + (i * 3 + k) * 2 + p] = acc[r];
                }
            }
        }
    }
}

extern "C" void kernel_launch(void* const* d_in, const int* in_sizes, int n_in,
                              void* d_out, int out_size, void* d_ws, size_t ws_size,
                              hipStream_t stream) {
    (void)in_sizes; (void)n_in; (void)out_size; (void)ws_size;
    const float* x = (const float*)d_in[0];
    const float* weight = (const float*)d_in[1];
    float* out = (float*)d_out;
    h16* wsA = (h16*)d_ws;   // 41*32*16 halves = 42 KB of packed W2 tiles

    lge_pack_w2<<<MT_G1, 32, 0, stream>>>(weight, wsA);
    lge_main<<<2 * XSITES, 32, 0, stream>>>(x, wsA, out);
}
